// CNNLSTM_22110491640589
// MI455X (gfx1250) — matrix operand order repaired and checked by reading, not compiled, not run on hardware
//
#include <hip/hip_runtime.h>
#include <math.h>

// ---------------- model constants ----------------
#define TC   512          // time steps after conv stack
#define BB   8            // batch
#define HID  768
#define G4   3072         // 4*HID
#define F0   1312         // conv feature size (32*41)
#define NCLS 29
#define EPSBN 1e-5f

typedef __attribute__((ext_vector_type(16))) __bf16 bf16x16;
typedef __attribute__((ext_vector_type(8)))  __bf16 bf16x8;
typedef __attribute__((ext_vector_type(8)))  float  f32x8;

// ---------------- WMMA helpers (CDNA5, wave32) ----------------
__device__ __forceinline__ f32x8 wmma_bf16(bf16x16 a, bf16x16 b, f32x8 c) {
  // D = A(16x32 bf16) x B(32x16 bf16) + C(16x16 f32)
  return __builtin_amdgcn_wmma_f32_16x16x32_bf16(false, a, false, b, (short)0, c,
                                                 false, false);
}

// A fragment: row-major A (rowstride elems). lane l: row = l&15, khalf = l>>4.
// a[i]   = A[m][kbase + khalf*16 + i]     (i=0..7)
// a[8+i] = A[m][kbase + khalf*16 + 8 + i]
__device__ __forceinline__ bf16x16 load_a_frag(const __bf16* A, int rowstride,
                                               int mrow, int kbase, int lane) {
  int m = lane & 15, kh = (lane >> 4) & 1;
  const __bf16* p = A + (size_t)(mrow + m) * rowstride + kbase + kh * 16;
  bf16x8 lo = *(const bf16x8*)(p);
  bf16x8 hi = *(const bf16x8*)(p + 8);
  bf16x16 r;
#pragma unroll
  for (int i = 0; i < 8; ++i) { r[i] = lo[i]; r[8 + i] = hi[i]; }
  return r;
}

// B fragment from BT (N x K row-major, BT[n][k] = B[k][n]).
// lane l: col n = l&15, khalf = l>>4; b[i] = B[kbase + khalf*16 + i][n]
__device__ __forceinline__ bf16x16 load_b_frag(const __bf16* BT, int rowstride,
                                               int nbase, int kbase, int lane) {
  int n = lane & 15, kh = (lane >> 4) & 1;
  return *(const bf16x16*)(BT + (size_t)(nbase + n) * rowstride + kbase + kh * 16);
}

__device__ __forceinline__ float sigf(float x) { return 1.f / (1.f + __expf(-x)); }
__device__ __forceinline__ int outlen_of(int len) { return (len - 1) / 2 + 1; }

// ================= conv stage =================

// conv1: in (8,1,161,1024) -> out (8,81,512,32), stride (2,2), pad (20,5), +bias, mask (keep d>=len)
__global__ void conv1_kernel(const float* __restrict__ in, const float* __restrict__ w,
                             const float* __restrict__ bias, const int* __restrict__ lengths,
                             float* __restrict__ out) {
  int idx = blockIdx.x * blockDim.x + threadIdx.x;  // over 8*81*512, t fastest
  if (idx >= 8 * 81 * 512) return;
  int t = idx % 512, d = (idx / 512) % 81, b = idx / (512 * 81);
  float* op = out + (size_t)idx * 32;
  if (d < lengths[b]) {  // reference mask keeps d >= lengths[b]
#pragma unroll
    for (int c = 0; c < 32; ++c) op[c] = 0.f;
    return;
  }
  float acc[32];
#pragma unroll
  for (int c = 0; c < 32; ++c) acc[c] = bias[c];
  for (int kh = 0; kh < 41; ++kh) {
    int dd = 2 * d + kh - 20;
    if ((unsigned)dd >= 161u) continue;
    const float* xrow = in + ((size_t)b * 161 + dd) * 1024;
#pragma unroll
    for (int kw = 0; kw < 11; ++kw) {
      int tt = 2 * t + kw - 5;
      if ((unsigned)tt >= 1024u) continue;
      float xv = xrow[tt];
      const float* wr = w + (kh * 11 + kw) * 32;
#pragma unroll
      for (int c = 0; c < 32; ++c) acc[c] = fmaf(xv, wr[c], acc[c]);
    }
  }
#pragma unroll
  for (int c = 0; c < 32; ++c) op[c] = acc[c];
}

// batch-stat BN reduction: one block per channel, deterministic tree reduce
__global__ void bnstats_kernel(const float* __restrict__ x, int nper, float* __restrict__ stats) {
  int c = blockIdx.x;  // 0..31
  float s = 0.f, s2 = 0.f;
  for (int i = threadIdx.x; i < nper; i += blockDim.x) {
    float v = x[(size_t)i * 32 + c];
    s += v; s2 += v * v;
  }
  __shared__ float sh[256], sh2[256];
  sh[threadIdx.x] = s; sh2[threadIdx.x] = s2;
  __syncthreads();
  for (int o = 128; o > 0; o >>= 1) {
    if (threadIdx.x < o) { sh[threadIdx.x] += sh[threadIdx.x + o]; sh2[threadIdx.x] += sh2[threadIdx.x + o]; }
    __syncthreads();
  }
  if (threadIdx.x == 0) {
    float mean = sh[0] / (float)nper;
    float var  = sh2[0] / (float)nper - mean * mean;
    stats[c] = mean; stats[32 + c] = var;
  }
}

// y = mask(d>=len) ? clip(bn(x),0,20) : 0, in place. layout (8, Ddim, 512, 32)
__global__ void bnapply_kernel(float* __restrict__ x, const float* __restrict__ stats,
                               const float* __restrict__ scale, const float* __restrict__ bias,
                               const int* __restrict__ lengths, int Ddim, long total) {
  for (long idx = (long)blockIdx.x * blockDim.x + threadIdx.x; idx < total;
       idx += (long)gridDim.x * blockDim.x) {
    int c = idx & 31;
    long r = idx >> 5;
    int d = (int)((r / 512) % Ddim);
    int b = (int)(r / (512L * Ddim));
    float v = x[idx];
    v = (v - stats[c]) * rsqrtf(stats[32 + c] + EPSBN) * scale[c] + bias[c];
    v = fminf(fmaxf(v, 0.f), 20.f);
    x[idx] = (d >= lengths[b]) ? v : 0.f;
  }
}

// conv2: in (8,81,512,32) -> out (8,41,512,32), stride (2,1), pad (10,5), +bias, mask
// grid (8*41, 4), block 128 (each thread one t), weight kh-slab staged in LDS
__global__ void conv2_kernel(const float* __restrict__ in, const float* __restrict__ w,
                             const float* __restrict__ bias, const int* __restrict__ lengths,
                             float* __restrict__ out) {
  __shared__ float wsh[11 * 32 * 32];
  int bd = blockIdx.x;
  int d = bd % 41, b = bd / 41;
  int t = blockIdx.y * 128 + threadIdx.x;
  float acc[32];
#pragma unroll
  for (int c = 0; c < 32; ++c) acc[c] = bias[c];
  for (int kh = 0; kh < 21; ++kh) {
    __syncthreads();
    for (int i = threadIdx.x; i < 11 * 32 * 32; i += 128) wsh[i] = w[kh * 11 * 32 * 32 + i];
    __syncthreads();
    int dd = 2 * d + kh - 10;
    if ((unsigned)dd >= 81u) continue;
#pragma unroll
    for (int kw = 0; kw < 11; ++kw) {
      int tt = t + kw - 5;
      if ((unsigned)tt >= 512u) continue;
      const float* xr = in + (((size_t)b * 81 + dd) * 512 + tt) * 32;
      const float* wr = wsh + kw * 32 * 32;
      for (int ci = 0; ci < 32; ++ci) {
        float xv = xr[ci];
#pragma unroll
        for (int c = 0; c < 32; ++c) acc[c] = fmaf(xv, wr[ci * 32 + c], acc[c]);
      }
    }
  }
  float* op = out + (((size_t)b * 41 + d) * 512 + t) * 32;
  bool keep = (d >= lengths[b]);
#pragma unroll
  for (int c = 0; c < 32; ++c) op[c] = keep ? acc[c] : 0.f;
}

// ================= LSTM stage =================

// transpose+convert weights: src (dirs, K, N) f32 -> dst (dirs, N, K) bf16
__global__ void wt_transpose_bf16(const float* __restrict__ src, __bf16* __restrict__ dst,
                                  int K, int N, long total) {
  for (long idx = (long)blockIdx.x * blockDim.x + threadIdx.x; idx < total;
       idx += (long)gridDim.x * blockDim.x) {
    int k = (int)(idx % K);
    long r = idx / K;
    int n = (int)(r % N);
    int dd = (int)(r / N);
    dst[idx] = (__bf16)src[((size_t)dd * K + k) * N + n];
  }
}

// layer0 prep: conv2 output (8,41,512,32) -> xF/xB (512,8,1312) bf16, feature = c*41+d,
// backward sequence rev[t] = x[(l-1-t) mod 512]
__global__ void prep0_kernel(const float* __restrict__ cv, const int* __restrict__ lengths,
                             __bf16* __restrict__ xF, __bf16* __restrict__ xB) {
  long total = (long)TC * BB * F0;
  for (long idx = (long)blockIdx.x * blockDim.x + threadIdx.x; idx < total;
       idx += (long)gridDim.x * blockDim.x) {
    int k = (int)(idx % F0);
    long tb = idx / F0;
    int b = (int)(tb % BB);
    int t = (int)(tb / BB);
    int c = k / 41, d = k % 41;
    xF[idx] = (__bf16)cv[(((size_t)b * 41 + d) * 512 + t) * 32 + c];
    int l = outlen_of(lengths[b]);
    int tr = (l - 1 - t) % TC; if (tr < 0) tr += TC;
    xB[idx] = (__bf16)cv[(((size_t)b * 41 + d) * 512 + tr) * 32 + c];
  }
}

// layers 1..4 prep: running BN then fwd/rev bf16 conversion; xin (512,8,768)
__global__ void prepL_kernel(const float* __restrict__ xin, const int* __restrict__ lengths,
                             const float* __restrict__ sc, const float* __restrict__ bi,
                             const float* __restrict__ mu, const float* __restrict__ var,
                             __bf16* __restrict__ xF, __bf16* __restrict__ xB) {
  long total = (long)TC * BB * HID;
  for (long idx = (long)blockIdx.x * blockDim.x + threadIdx.x; idx < total;
       idx += (long)gridDim.x * blockDim.x) {
    int j = (int)(idx % HID);
    long tb = idx / HID;
    int b = (int)(tb % BB);
    int t = (int)(tb / BB);
    float rs = rsqrtf(var[j] + EPSBN) * sc[j];
    xF[idx] = (__bf16)((xin[idx] - mu[j]) * rs + bi[j]);
    int l = outlen_of(lengths[b]);
    int tr = (l - 1 - t) % TC; if (tr < 0) tr += TC;
    float vr = xin[((size_t)tr * BB + b) * HID + j];
    xB[idx] = (__bf16)((vr - mu[j]) * rs + bi[j]);
  }
}

// input projection: xw[dir][t*8+b][n] = sum_k x[dir][t][b][k]*Wi[k][n] + bias[n]
// grid (256 Mtiles, 12 Nchunks, 2 dirs); block 256 = 8 waves; wave does 2 16x16 tiles
__global__ void __launch_bounds__(256)
lstm_inproj(const __bf16* __restrict__ xF, const __bf16* __restrict__ xB,
            const __bf16* __restrict__ wT, const float* __restrict__ bias,
            float* __restrict__ xw, int K) {
  int dir = blockIdx.z;
  int wv = threadIdx.x >> 5, lane = threadIdx.x & 31;
  const __bf16* X = dir ? xB : xF;
  const __bf16* W = wT + (size_t)dir * G4 * K;
  int mrow = blockIdx.x * 16;
  int n0 = blockIdx.y * 256 + wv * 32;
  f32x8 c0 = {}, c1 = {};
  for (int kb = 0; kb < K; kb += 32) {
    bf16x16 a  = load_a_frag(X, K, mrow, kb, lane);
    bf16x16 b0 = load_b_frag(W, K, n0, kb, lane);
    bf16x16 b1 = load_b_frag(W, K, n0 + 16, kb, lane);
    c0 = wmma_bf16(a, b0, c0);
    c1 = wmma_bf16(a, b1, c1);
  }
  const float* bs = bias + (size_t)dir * G4;
  float* out = xw + (size_t)dir * (TC * BB) * G4;
  int row0 = mrow + ((lane >> 4) & 1) * 8;
  int col0 = n0 + (lane & 15);
#pragma unroll
  for (int r = 0; r < 8; ++r) {
    out[(size_t)(row0 + r) * G4 + col0]      = c0[r] + bs[col0];
    out[(size_t)(row0 + r) * G4 + col0 + 16] = c1[r] + bs[col0 + 16];
  }
}

// one recurrent step: gates = xw[t] + h_prev @ Wh, then fused LSTM cell.
// grid (12 hidden tiles of 64, 2 dirs); block 256 = 8 waves; wave does 2 of 16 N-tiles
__global__ void __launch_bounds__(256)
lstm_step(int t, const float* __restrict__ xw, const __bf16* __restrict__ whT,
          __bf16* __restrict__ hbf, float* __restrict__ cst,
          float* __restrict__ hseqF, float* __restrict__ hseqB) {
  __shared__ float G[16 * 256];
  int dir = blockIdx.y;
  int j0 = blockIdx.x * 64;
  int wv = threadIdx.x >> 5, lane = threadIdx.x & 31;
  const __bf16* A = hbf + (size_t)dir * 16 * HID;     // 16 x 768 (rows 8..15 zero)
  const __bf16* W = whT + (size_t)dir * G4 * HID;     // (3072, 768) = Wh^T
  int tn0 = wv * 2;
  int g0 = tn0 >> 2, st0 = tn0 & 3;
  int g1 = (tn0 + 1) >> 2, st1 = (tn0 + 1) & 3;
  int nb0 = g0 * HID + j0 + st0 * 16;
  int nb1 = g1 * HID + j0 + st1 * 16;
  f32x8 c0 = {}, c1 = {};
  for (int kb = 0; kb < HID; kb += 32) {
    bf16x16 a  = load_a_frag(A, HID, 0, kb, lane);
    bf16x16 b0 = load_b_frag(W, HID, nb0, kb, lane);
    bf16x16 b1 = load_b_frag(W, HID, nb1, kb, lane);
    c0 = wmma_bf16(a, b0, c0);
    c1 = wmma_bf16(a, b1, c1);
  }
  int row0 = ((lane >> 4) & 1) * 8;
  int c16 = lane & 15;
#pragma unroll
  for (int r = 0; r < 8; ++r) {
    G[(row0 + r) * 256 + tn0 * 16 + c16]       = c0[r];   // lds col = g*64 + jj
    G[(row0 + r) * 256 + (tn0 + 1) * 16 + c16] = c1[r];
  }
  __syncthreads();

  float* c_state = cst + (size_t)dir * BB * HID;
  float* hseq = dir ? hseqB : hseqF;
  __bf16* hb = hbf + (size_t)dir * 16 * HID;
  const float* xwrow = xw + (size_t)dir * (TC * BB) * G4 + (size_t)t * BB * G4;
  for (int p = threadIdx.x; p < BB * 64; p += 256) {
    int b = p >> 6, jj = p & 63;
    int j = j0 + jj;
    const float* xr = xwrow + (size_t)b * G4;
    float gi = G[b * 256 + 0 * 64 + jj] + xr[0 * HID + j];
    float gf = G[b * 256 + 1 * 64 + jj] + xr[1 * HID + j];
    float gg = G[b * 256 + 2 * 64 + jj] + xr[2 * HID + j];
    float go = G[b * 256 + 3 * 64 + jj] + xr[3 * HID + j];
    float cn = sigf(gf) * c_state[b * HID + j] + sigf(gi) * tanhf(gg);
    c_state[b * HID + j] = cn;
    float h = sigf(go) * tanhf(cn);
    hseq[((size_t)t * BB + b) * HID + j] = h;
    hb[b * HID + j] = (__bf16)h;
  }
}

// out[t][b][j] = fwd[t][b][j] + bwd[(l-1-t) mod 512][b][j]  (flip-back + dir sum)
__global__ void combine_kernel(const float* __restrict__ hF, const float* __restrict__ hB,
                               const int* __restrict__ lengths, float* __restrict__ out) {
  long total = (long)TC * BB * HID;
  for (long idx = (long)blockIdx.x * blockDim.x + threadIdx.x; idx < total;
       idx += (long)gridDim.x * blockDim.x) {
    int j = (int)(idx % HID);
    long tb = idx / HID;
    int b = (int)(tb % BB);
    int t = (int)(tb / BB);
    int l = outlen_of(lengths[b]);
    int tr = (l - 1 - t) % TC; if (tr < 0) tr += TC;
    out[idx] = hF[idx] + hB[((size_t)tr * BB + b) * HID + j];
  }
}

// final: running BN + FC(768x29) + log_softmax, transpose to (B, T, 29). block per (t,b).
__global__ void fc_kernel(const float* __restrict__ x, const float* __restrict__ sc,
                          const float* __restrict__ bi, const float* __restrict__ mu,
                          const float* __restrict__ var, const float* __restrict__ fw,
                          float* __restrict__ out) {
  __shared__ float xs[HID];
  __shared__ float lg[NCLS];
  __shared__ float red[2];
  int tb = blockIdx.x;
  int b = tb % BB, t = tb / BB;
  for (int j = threadIdx.x; j < HID; j += 64) {
    float v = x[((size_t)t * BB + b) * HID + j];
    xs[j] = (v - mu[j]) * rsqrtf(var[j] + EPSBN) * sc[j] + bi[j];
  }
  __syncthreads();
  int c = threadIdx.x;
  if (c < NCLS) {
    float a = 0.f;
    for (int j = 0; j < HID; ++j) a = fmaf(xs[j], fw[j * NCLS + c], a);
    lg[c] = a;
  }
  __syncthreads();
  if (threadIdx.x == 0) {
    float m = -1e30f;
    for (int i = 0; i < NCLS; ++i) m = fmaxf(m, lg[i]);
    float s = 0.f;
    for (int i = 0; i < NCLS; ++i) s += __expf(lg[i] - m);
    red[0] = m; red[1] = logf(s);
  }
  __syncthreads();
  if (c < NCLS) out[((size_t)b * TC + t) * NCLS + c] = lg[c] - red[0] - red[1];
}

__global__ void outlen_kernel(const int* __restrict__ lengths, float* __restrict__ out) {
  int b = threadIdx.x;
  if (b < BB) out[b] = (float)outlen_of(lengths[b]);
}

// ================= host driver =================
extern "C" void kernel_launch(void* const* d_in, const int* in_sizes, int n_in,
                              void* d_out, int out_size, void* d_ws, size_t ws_size,
                              hipStream_t stream) {
  (void)in_sizes; (void)n_in; (void)out_size; (void)ws_size;
  const float* inputs    = (const float*)d_in[0];
  const int*   lengths   = (const int*)d_in[1];
  const float* conv1_w   = (const float*)d_in[2];
  const float* conv1_b   = (const float*)d_in[3];
  const float* bn1_scale = (const float*)d_in[4];
  const float* bn1_bias  = (const float*)d_in[5];
  const float* conv2_w   = (const float*)d_in[6];
  const float* conv2_b   = (const float*)d_in[7];
  const float* bn2_scale = (const float*)d_in[8];
  const float* bn2_bias  = (const float*)d_in[9];
  const float* lstm0_wi  = (const float*)d_in[10];  // (2, F0, 3072)
  const float* lstm0_wh  = (const float*)d_in[11];  // (2, 768, 3072)
  const float* lstm0_b   = (const float*)d_in[12];  // (2, 3072)
  const float* lstm_wi   = (const float*)d_in[13];  // (4, 2, 768, 3072)
  const float* lstm_wh   = (const float*)d_in[14];
  const float* lstm_b    = (const float*)d_in[15];  // (4, 2, 3072)
  const float* rnn_bn_sc = (const float*)d_in[16];  // (4, 768)
  const float* rnn_bn_bi = (const float*)d_in[17];
  const float* rnn_bn_mu = (const float*)d_in[18];
  const float* rnn_bn_va = (const float*)d_in[19];
  const float* fc_bn_sc  = (const float*)d_in[20];
  const float* fc_bn_bi  = (const float*)d_in[21];
  const float* fc_bn_mu  = (const float*)d_in[22];
  const float* fc_bn_va  = (const float*)d_in[23];
  const float* fc_w      = (const float*)d_in[24];  // (768, 29)
  float* out = (float*)d_out;

  char* ws = (char*)d_ws;
  // xw region (100.66 MB) is reused: conv buffers alias inside it (conv phase ends
  // before the first xw write in lstm_inproj).
  const size_t XW_BYTES   = (size_t)2 * TC * BB * G4 * 4;           // 100,663,296
  float*  conv1buf = (float*)(ws + 0);                               // 42.47 MB
  float*  conv2buf = (float*)(ws + (size_t)8 * 81 * 512 * 32 * 4);   // 21.50 MB
  float*  xw       = (float*)(ws + 0);
  size_t off = XW_BYTES;
  __bf16* wtWi = (__bf16*)(ws + off); off += (size_t)2 * G4 * F0 * 2;    // max K=1312
  __bf16* wtWh = (__bf16*)(ws + off); off += (size_t)2 * G4 * HID * 2;
  __bf16* xbfF = (__bf16*)(ws + off); off += (size_t)TC * BB * F0 * 2;
  __bf16* xbfB = (__bf16*)(ws + off); off += (size_t)TC * BB * F0 * 2;
  float*  hseqF = (float*)(ws + off); off += (size_t)TC * BB * HID * 4;
  float*  hseqB = (float*)(ws + off); off += (size_t)TC * BB * HID * 4;
  float*  bufA  = (float*)(ws + off); off += (size_t)TC * BB * HID * 4;
  float*  bufB  = (float*)(ws + off); off += (size_t)TC * BB * HID * 4;
  __bf16* hbf   = (__bf16*)(ws + off); size_t hbf_off = off; off += (size_t)2 * 16 * HID * 2;
  float*  cst   = (float*)(ws + off);  off += (size_t)2 * BB * HID * 4;
  float*  stats = (float*)(ws + off);  off += 256;
  const size_t HC_BYTES = (size_t)2 * 16 * HID * 2 + (size_t)2 * BB * HID * 4;

  // ---- conv stage ----
  conv1_kernel<<<(8 * 81 * 512 + 127) / 128, 128, 0, stream>>>(inputs, conv1_w, conv1_b,
                                                               lengths, conv1buf);
  bnstats_kernel<<<32, 256, 0, stream>>>(conv1buf, 8 * 81 * 512, stats);
  bnapply_kernel<<<4096, 256, 0, stream>>>(conv1buf, stats, bn1_scale, bn1_bias, lengths,
                                           81, (long)8 * 81 * 512 * 32);
  conv2_kernel<<<dim3(8 * 41, 4), 128, 0, stream>>>(conv1buf, conv2_w, conv2_b, lengths,
                                                    conv2buf);
  bnstats_kernel<<<32, 256, 0, stream>>>(conv2buf, 8 * 41 * 512, stats);
  bnapply_kernel<<<4096, 256, 0, stream>>>(conv2buf, stats, bn2_scale, bn2_bias, lengths,
                                           41, (long)8 * 41 * 512 * 32);

  // ---- 5 BiLSTM layers ----
  float* cur = bufA;
  float* nxt = bufB;
  for (int L = 0; L < 5; ++L) {
    const float* wiSrc; const float* whSrc; const float* biasSrc; int K;
    if (L == 0) {
      wiSrc = lstm0_wi; whSrc = lstm0_wh; biasSrc = lstm0_b; K = F0;
      prep0_kernel<<<8192, 256, 0, stream>>>(conv2buf, lengths, xbfF, xbfB);
    } else {
      int i = L - 1;
      wiSrc = lstm_wi + (size_t)i * 2 * HID * G4;
      whSrc = lstm_wh + (size_t)i * 2 * HID * G4;
      biasSrc = lstm_b + (size_t)i * 2 * G4;
      K = HID;
      prepL_kernel<<<8192, 256, 0, stream>>>(cur, lengths,
                                             rnn_bn_sc + i * HID, rnn_bn_bi + i * HID,
                                             rnn_bn_mu + i * HID, rnn_bn_va + i * HID,
                                             xbfF, xbfB);
    }
    long twi = (long)2 * G4 * K;
    wt_transpose_bf16<<<8192, 256, 0, stream>>>(wiSrc, wtWi, K, G4, twi);
    wt_transpose_bf16<<<8192, 256, 0, stream>>>(whSrc, wtWh, HID, G4, (long)2 * G4 * HID);
    hipMemsetAsync(ws + hbf_off, 0, HC_BYTES, stream);  // h0 = c0 = 0 (+ zero pad rows)

    lstm_inproj<<<dim3(TC * BB / 16, G4 / 256, 2), 256, 0, stream>>>(xbfF, xbfB, wtWi,
                                                                     biasSrc, xw, K);
    for (int t = 0; t < TC; ++t)
      lstm_step<<<dim3(HID / 64, 2), 256, 0, stream>>>(t, xw, wtWh, hbf, cst, hseqF, hseqB);
    combine_kernel<<<8192, 256, 0, stream>>>(hseqF, hseqB, lengths, nxt);
    float* tmp = cur; cur = nxt; nxt = tmp;
  }

  // ---- FC + log_softmax + out_len ----
  fc_kernel<<<TC * BB, 64, 0, stream>>>(cur, fc_bn_sc, fc_bn_bi, fc_bn_mu, fc_bn_va,
                                        fc_w, out);
  outlen_kernel<<<1, BB, 0, stream>>>(lengths, out + (size_t)BB * TC * NCLS);
}
